// Quardrant_Attention_Layer_67070209294593
// MI455X (gfx1250) — matrix operand order repaired and checked by reading, compiled, not run on hardware
//
#include <hip/hip_runtime.h>
#include <hip/hip_bf16.h>

// ---------------------------------------------------------------------------
// Quadrant attention, fused single kernel for MI455X (gfx1250, wave32).
//   B=4, C=64, H=W=160, NH=4  ->  N = B*H*W... (raw reshape) = 102400 rows.
// Strategy: f16 WMMA (16x16x32, f32 accum) for theta/phi/psi GEMMs,
// f32 softmax via 4-lane shuffles, final nh-sum via 2-lane shuffles.
// Memory-bound target: ~52 MB HBM traffic @ 23.3 TB/s ~ 2.3 us.
// ---------------------------------------------------------------------------

typedef __attribute__((ext_vector_type(16))) _Float16 v16h;
typedef __attribute__((ext_vector_type(8)))  _Float16 v8h;
typedef __attribute__((ext_vector_type(8)))  float    v8f;
typedef __attribute__((ext_vector_type(4)))  float    v4f;

#define LDS_WTH   0         // 256x64 f16 = 32768 B
#define LDS_WPH   32768     // 256x64 f16 = 32768 B
#define LDS_WPS   65536     //  64x64 f16 =  8192 B
#define LDS_PHI   73728     // 8 waves * [4][16][64] f16 = 65536 B
#define LDS_TOTAL (73728 + 8 * 8192)   // 139264 B dynamic LDS

__device__ __forceinline__ v8f wmma16(v16h a, v16h b, v8f c) {
  // D = A(16x32 f16) x B(32x16 f16) + C(16x16 f32)
  return __builtin_amdgcn_wmma_f32_16x16x32_f16(false, a, false, b,
                                                (short)0, c, false, false);
}

// A-fragment from a global f32 row.  Lane layout (contiguous K order):
// lane<16 : row=lane,  elems 0..15 = K(base..base+15)
// lane>=16: row=l-16,  same with base += 16  (caller passes col0 = base)
__device__ __forceinline__ v16h load_a_f32(const float* __restrict__ rowp, int col0) {
  v4f f0 = *(const v4f*)(rowp + col0);
  v4f f1 = *(const v4f*)(rowp + col0 + 4);
  v4f f2 = *(const v4f*)(rowp + col0 + 8);
  v4f f3 = *(const v4f*)(rowp + col0 + 12);
  v16h a;
#pragma unroll
  for (int i = 0; i < 4; ++i) {
    a[i]      = (_Float16)f0[i];
    a[4 + i]  = (_Float16)f1[i];
    a[8 + i]  = (_Float16)f2[i];
    a[12 + i] = (_Float16)f3[i];
  }
  return a;
}

// A-fragment from an f16 LDS row (same layout); p points at row + half*16.
__device__ __forceinline__ v16h load_a_h16(const _Float16* p, int off) {
  v8h lo = *(const v8h*)(p + off);
  v8h hi = *(const v8h*)(p + off + 8);
  v16h a;
#pragma unroll
  for (int i = 0; i < 8; ++i) { a[i] = lo[i]; a[8 + i] = hi[i]; }
  return a;
}

__device__ __forceinline__ unsigned pack2(float lo, float hi) {
  union { _Float16 h[2]; unsigned u; } p;
  p.h[0] = (_Float16)lo; p.h[1] = (_Float16)hi;
  return p.u;
}

__global__ void __launch_bounds__(256, 1)
qattn_kernel(const float* __restrict__ x,
             const float* __restrict__ Wth_g,   // [256,64] row-major
             const float* __restrict__ Wph_g,   // [256,64]
             const float* __restrict__ Wps_g,   // [64,64]
             const float* __restrict__ gamma_p,
             float* __restrict__ out)
{
  extern __shared__ __attribute__((aligned(128))) char smem[];
  _Float16* wth  = (_Float16*)(smem + LDS_WTH);   // B-col j of W_theta^T == row j
  _Float16* wph  = (_Float16*)(smem + LDS_WPH);
  _Float16* wps  = (_Float16*)(smem + LDS_WPS);

  const int tid = threadIdx.x;

  // --- stage f32 weights into LDS as f16 (rows contiguous == B columns) ---
  {
    const float2* s0 = (const float2*)Wth_g;  unsigned* d0 = (unsigned*)wth;
    const float2* s1 = (const float2*)Wph_g;  unsigned* d1 = (unsigned*)wph;
    const float2* s2 = (const float2*)Wps_g;  unsigned* d2 = (unsigned*)wps;
    for (int i = tid; i < 8192; i += 256) { float2 f = s0[i]; d0[i] = pack2(f.x, f.y); }
    for (int i = tid; i < 8192; i += 256) { float2 f = s1[i]; d1[i] = pack2(f.x, f.y); }
    for (int i = tid; i < 2048; i += 256) { float2 f = s2[i]; d2[i] = pack2(f.x, f.y); }
  }
  __syncthreads();

  const int lane = tid & 31;
  const int wv   = tid >> 5;
  const int n16  = lane & 15;
  const int half = lane >> 4;      // K-half selector (A/B) / row-half (C,D)
  const int q    = n16 >> 2;       // c&3 within a j-tile
  const int nh   = n16 & 3;        // head index carried by this lane
  _Float16* phiw = (_Float16*)(smem + LDS_PHI + wv * 8192);  // [4][16][64] f16

  const int   Rbase = blockIdx.x * 128 + wv * 16;            // 16 rows per wave
  const float* xrow = x + (size_t)(Rbase + n16) * 64;
  const float gamma = gamma_p[0];

  // A fragments of this wave's 16x64 x-tile (K split 0..31 / 32..63)
  const v16h a0 = load_a_f32(xrow, half * 16);
  const v16h a1 = load_a_f32(xrow, 32 + half * 16);

  // e[jt][v] = attn logits (later softmax probs).  Lane n,VGPR v holds
  // (row = v+8*half, c = 4*jt+q, head = nh).
  float e[16][8];

#pragma unroll
  for (int jt = 0; jt < 16; ++jt) {
    const _Float16* bt = wth + (jt * 16 + n16) * 64 + half * 16;
    const _Float16* bp = wph + (jt * 16 + n16) * 64 + half * 16;
    v8f dt = {}; v8f dp = {};
    dt = wmma16(a0, *(const v16h*)(bt),      dt);
    dt = wmma16(a1, *(const v16h*)(bt + 32), dt);
    dp = wmma16(a0, *(const v16h*)(bp),      dp);
    dp = wmma16(a1, *(const v16h*)(bp + 32), dp);
#pragma unroll
    for (int v = 0; v < 8; ++v) e[jt][v] = dt[v] * dp[v] * 0.0625f; // /(C/groups)
    // scatter phi (f16) as Phi_g[m][c] = phi[m, 4c+g] for the psi GEMM A-matrix
    _Float16* pd = phiw + (nh * 16) * 64 + (4 * jt + q);
#pragma unroll
    for (int v = 0; v < 8; ++v) pd[(v + 8 * half) * 64] = (_Float16)dp[v];
  }

  // --- softmax over c (64 values = 16 local jt x 4 lanes via xor 4,8) ---
#pragma unroll
  for (int v = 0; v < 8; ++v) {
    float m = e[0][v];
#pragma unroll
    for (int jt = 1; jt < 16; ++jt) m = fmaxf(m, e[jt][v]);
    m = fmaxf(m, __shfl_xor(m, 4, 32));
    m = fmaxf(m, __shfl_xor(m, 8, 32));
    float s = 0.f;
#pragma unroll
    for (int jt = 0; jt < 16; ++jt) { float ex = __expf(e[jt][v] - m); e[jt][v] = ex; s += ex; }
    s += __shfl_xor(s, 4, 32);
    s += __shfl_xor(s, 8, 32);
    const float inv = 1.0f / s;
#pragma unroll
    for (int jt = 0; jt < 16; ++jt) e[jt][v] *= inv;
  }

  // --- psi GEMMs + fused combine.  psi tile (g,ct) element on lane n,VGPR v
  // pairs with prob e[4g+ct][v] on the SAME lane (c=16g+4ct+q, head=nh). ---
#pragma unroll
  for (int g = 0; g < 4; ++g) {
    const _Float16* par = phiw + (g * 16 + n16) * 64 + half * 16;
    const v16h pa0 = load_a_h16(par, 0);
    const v16h pa1 = load_a_h16(par, 32);
#pragma unroll
    for (int ct = 0; ct < 4; ++ct) {
      const _Float16* bs = wps + (ct * 16 + n16) * 64 + half * 16;
      v8f d = {};
      d = wmma16(pa0, *(const v16h*)(bs),      d);
      d = wmma16(pa1, *(const v16h*)(bs + 32), d);
      const int jt2 = 4 * g + ct;
      const int c   = 16 * g + 4 * ct + q;
#pragma unroll
      for (int v = 0; v < 8; ++v) {
        float t = e[jt2][v] * d[v];
        t += __shfl_xor(t, 1, 32);      // sum over heads (nh bits = lane 0..1)
        t += __shfl_xor(t, 2, 32);
        if ((v >> 1) == nh) {           // 4 duplicate lanes -> split rows by nh
          const size_t addr = (size_t)(Rbase + v + 8 * half) * 64 + c;
          out[addr] = gamma * t + x[addr];
        }
      }
    }
  }
}

extern "C" void kernel_launch(void* const* d_in, const int* in_sizes, int n_in,
                              void* d_out, int out_size, void* d_ws, size_t ws_size,
                              hipStream_t stream) {
  (void)n_in; (void)d_ws; (void)ws_size; (void)out_size;
  const float* x   = (const float*)d_in[0];
  const float* Wt  = (const float*)d_in[1];
  const float* Wp  = (const float*)d_in[2];
  const float* Ws  = (const float*)d_in[3];
  const float* gam = (const float*)d_in[4];
  float* out = (float*)d_out;

  const int N = in_sizes[0] / 64;          // 102400 rows
  dim3 grid(N / 128), block(256);          // 8 waves * 16 rows = 128 rows/block
  hipLaunchKernelGGL(qattn_kernel, grid, block, LDS_TOTAL, stream,
                     x, Wt, Wp, Ws, gam, out);
}
